// PotNet_20633022890686
// MI455X (gfx1250) — matrix operand order repaired and checked by reading, not compiled, not run on hardware
//
#include <hip/hip_runtime.h>

typedef __bf16 bf16;
typedef __attribute__((ext_vector_type(16))) __bf16 v16bf;
typedef __attribute__((ext_vector_type(8)))  __bf16 v8bf;
typedef __attribute__((ext_vector_type(8)))  float  v8f;

static constexpr int cN   = 10000;
static constexpr int cE   = 200000;
static constexpr int cEP  = 200064;   // cE  padded to 128
static constexpr int cEI  = 100000;
static constexpr int cEIP = 100096;   // cEI padded to 128
static constexpr int cET  = 300000;
static constexpr int cETP = 300032;   // cET padded to 128 (128*2344)
static constexpr int cF   = 256;
static constexpr int cDIN = 92;
static constexpr int cIB  = 64;
static constexpr int cG   = 128;
static constexpr int cL   = 3;
static constexpr int cK3  = 768;

__device__ __forceinline__ void atomAddF(float* p, float v) {
  __hip_atomic_fetch_add(p, v, __ATOMIC_RELAXED, __HIP_MEMORY_SCOPE_AGENT);
}
__device__ __forceinline__ float dev_sigmoid(float x) { return 1.f / (1.f + __expf(-x)); }
__device__ __forceinline__ float dev_softplus(float x) { return (x > 20.f) ? x : log1pf(__expf(x)); }

// ---------------------------------------------------------------------------
// bf16 WMMA GEMM: out[M x 256] = act(A[M x K] @ Wp + bias), M % 128 == 0.
// Wp packed [K/32][256][32] bf16 (16 contiguous K per column per lane-half ->
// one 32B load per B operand). A operand matches B's K order:
// lane-half khalf holds K = {16*khalf..+15}.
// Block: 256 thr = 8 waves; tile 128(M) x 256(N).
// Wave: 32 rows x 128 cols = 16 accumulators; each B operand feeds 2 WMMAs
// from registers -> 1.25 VMEM loads per WMMA.
// ---------------------------------------------------------------------------
__global__ __launch_bounds__(256)
void k_gemm_bf16(const bf16* __restrict__ A, const bf16* __restrict__ Wp,
                 const float* __restrict__ bias, float* __restrict__ outF,
                 bf16* __restrict__ outB, int K, int act)
{
  const int tid    = threadIdx.x;
  const int wave   = tid >> 5;
  const int lane   = tid & 31;
  const int lane16 = lane & 15;
  const int khalf  = lane >> 4;
  const int rg     = wave & 3;            // 4 row groups of 32
  const int ncolB  = (wave >> 2) * 128;   // 2 col halves of 128

  const long rowBase = (long)blockIdx.x * 128 + (long)rg * 32;

  v8f acc[16] = {};                       // acc[s*8+t]: s = 16-row subtile, t = col tile
  const bf16* aRow0 = A + (rowBase + lane16) * (long)K;
  const bf16* aRow1 = aRow0 + 16 * (long)K;
  const int kSteps = K >> 5;

  for (int ks = 0; ks < kSteps; ++ks) {
    const long aOff = (long)ks * 32 + khalf * 16;
    v8bf a0l = *(const v8bf*)(aRow0 + aOff);
    v8bf a0h = *(const v8bf*)(aRow0 + aOff + 8);
    v8bf a1l = *(const v8bf*)(aRow1 + aOff);
    v8bf a1h = *(const v8bf*)(aRow1 + aOff + 8);
    v16bf a0 = __builtin_shufflevector(a0l, a0h, 0,1,2,3,4,5,6,7,8,9,10,11,12,13,14,15);
    v16bf a1 = __builtin_shufflevector(a1l, a1h, 0,1,2,3,4,5,6,7,8,9,10,11,12,13,14,15);
    const bf16* bBase = Wp + ((long)ks * 256 + ncolB + lane16) * 32 + khalf * 16;
#pragma unroll
    for (int t = 0; t < 8; ++t) {
      v16bf b = *(const v16bf*)(bBase + t * (16 * 32));
      acc[t]     = __builtin_amdgcn_wmma_f32_16x16x32_bf16(
                       false, a0, false, b, (short)0, acc[t],     false, false);
      acc[8 + t] = __builtin_amdgcn_wmma_f32_16x16x32_bf16(
                       false, a1, false, b, (short)0, acc[8 + t], false, false);
    }
  }

#pragma unroll
  for (int s = 0; s < 2; ++s) {
#pragma unroll
    for (int t = 0; t < 8; ++t) {
      const int col = ncolB + t * 16 + lane16;
      const float bv = bias ? bias[col] : 0.f;
#pragma unroll
      for (int r = 0; r < 8; ++r) {
        const long row = rowBase + s * 16 + r + khalf * 8;  // C layout: VGPR r -> M = 8*khalf + r
        float v = acc[s * 8 + t][r] + bv;
        if (act == 1)      v = v * dev_sigmoid(v);   // SiLU
        else if (act == 2) v = dev_softplus(v);      // softplus
        const long idx = row * 256 + col;
        if (outF) outF[idx] = v;
        else      outB[idx] = (bf16)v;
      }
    }
  }
}

// Pack W[K x 256] f32 -> Wp[K/32][256][32] bf16
__global__ void k_pack_w(const float* __restrict__ W, bf16* __restrict__ Wp, int K) {
  long i = (long)blockIdx.x * 256 + threadIdx.x;
  if (i >= (long)K * 256) return;
  int k = (int)(i / 256), n = (int)(i % 256);
  Wp[((long)(k >> 5) * 256 + n) * 32 + (k & 31)] = (bf16)W[(long)k * 256 + n];
}

// h = x @ atom_w + atom_b  (tiny GEMM, VALU is fine); also emit bf16 copy
__global__ void k_embed(const float* __restrict__ x, const float* __restrict__ w,
                        const float* __restrict__ b, float* __restrict__ h,
                        bf16* __restrict__ hb) {
  long i = (long)blockIdx.x * 256 + threadIdx.x;
  if (i >= (long)cN * cF) return;
  int n = (int)(i / cF), f = (int)(i % cF);
  float acc = b[f];
  for (int k = 0; k < cDIN; ++k) acc += x[(long)n * cDIN + k] * w[(long)k * cF + f];
  h[i] = acc; hb[i] = (bf16)acc;
}

// gaussian RBF of -0.75/d over 256 bins in [-4,4] -> bf16
__global__ void k_rbf_local(const float* __restrict__ ea, bf16* __restrict__ o) {
  long i = (long)blockIdx.x * 256 + threadIdx.x;
  if (i >= (long)cE * cF) return;
  int e = (int)(i / cF), bin = (int)(i % cF);
  float d = -0.75f / ea[e];
  float c = -4.f + 8.f * bin / 255.f;
  float t = 31.875f * (d - c);          // gamma = 255/8
  o[i] = (bf16)__expf(-t * t);
}

// multiquadric RBF of (inf_edge_attr @ POT) over 64 bins -> bf16
__global__ void k_rbf_inf(const float* __restrict__ a, bf16* __restrict__ o) {
  long i = (long)blockIdx.x * 256 + threadIdx.x;
  if (i >= (long)cEI * cIB) return;
  int e = (int)(i / cIB), bin = (int)(i % cIB);
  float d = a[e * 3] * 1.0f + a[e * 3 + 1] * 0.5f + a[e * 3 + 2] * 0.3f;
  float c = -4.f + 8.f * bin / 63.f;
  float t = 7.875f * (d - c);           // gamma = 63/8
  o[i] = (bf16)sqrtf(1.f + t * t);
}

__global__ void k_zero(float* p, long n) {
  long i = (long)blockIdx.x * 256 + threadIdx.x;
  if (i < n) p[i] = 0.f;
}

// per-column sum / sumsq over M rows of X[M x 256] (coalesced, atomics at end)
__global__ __launch_bounds__(256)
void k_colstats(const float* __restrict__ X, int M, int rowsPerBlk,
                float* __restrict__ sum, float* __restrict__ sumsq) {
  int col = threadIdx.x;
  long r0 = (long)blockIdx.x * rowsPerBlk;
  long r1 = r0 + rowsPerBlk; if (r1 > M) r1 = M;
  float s = 0.f, q = 0.f;
  for (long r = r0; r < r1; ++r) { float v = X[r * 256 + col]; s += v; q += v * v; }
  atomAddF(&sum[col], s); atomAddF(&sumsq[col], q);
}

// scale/shift for BN: y = scale*x + shift
__global__ void k_bn_finalize(const float* __restrict__ sum, const float* __restrict__ sumsq,
                              const float* __restrict__ g, const float* __restrict__ b,
                              int M, float* __restrict__ scale, float* __restrict__ shift) {
  int c = threadIdx.x;
  float mean = sum[c] / (float)M;
  float var  = sumsq[c] / (float)M - mean * mean;
  float sc   = g[c] * rsqrtf(var + 1e-5f);
  scale[c] = sc; shift[c] = b[c] - mean * sc;
}

// apply BN to f32 X and write bf16 (used for ef_inf rows of ef_bf)
__global__ void k_bn_apply_bf(const float* __restrict__ X, const float* __restrict__ scale,
                              const float* __restrict__ shift, bf16* __restrict__ o, long rows) {
  long i = (long)blockIdx.x * 256 + threadIdx.x;
  if (i >= rows * cF) return;
  int c = (int)(i % cF);
  o[i] = (bf16)(scale[c] * X[i] + shift[c]);
}

// z[e] = [ h_bf[tgt[e]] | h_bf[src[e]] | ef_bf[e] ]  (16B chunks)
__global__ void k_gather_z(const bf16* __restrict__ hb, const bf16* __restrict__ efb,
                           const int* __restrict__ eix, const int* __restrict__ iix,
                           bf16* __restrict__ z) {
  long i = (long)blockIdx.x * 256 + threadIdx.x;   // cET * 96 chunks of 16B
  if (i >= (long)cET * 96) return;
  int e = (int)(i / 96), p = (int)(i % 96);
  int seg = p >> 5, off = p & 31;                  // 32 x uint4 = 256 bf16 / segment
  int tgt = (e < cE) ? eix[cE + e] : iix[cEI + (e - cE)];
  int src = (e < cE) ? eix[e]       : iix[e - cE];
  const uint4* sp;
  if (seg == 0)      sp = (const uint4*)(hb + (long)tgt * cF) + off;
  else if (seg == 1) sp = (const uint4*)(hb + (long)src * cF) + off;
  else               sp = (const uint4*)(efb + (long)e * cF) + off;
  ((uint4*)(z + (long)e * cK3))[p] = *sp;
}

// msg_final = sigmoid(BN(gate)) * msg ; atomic segment-sum into agg[tgt]
__global__ void k_gate_agg(const float* __restrict__ gate, const float* __restrict__ msg,
                           const float* __restrict__ scale, const float* __restrict__ shift,
                           const int* __restrict__ eix, const int* __restrict__ iix,
                           float* __restrict__ agg) {
  long i = (long)blockIdx.x * 256 + threadIdx.x;
  if (i >= (long)cET * cF) return;
  int e = (int)(i / cF), c = (int)(i % cF);
  float gv = dev_sigmoid(scale[c] * gate[i] + shift[c]);
  int tgt = (e < cE) ? eix[cE + e] : iix[cEI + (e - cE)];
  atomAddF(&agg[(long)tgt * cF + c], gv * msg[i]);
}

// h = relu(h + BN(agg)); refresh bf16 copy
__global__ void k_update_h(float* __restrict__ h, const float* __restrict__ agg,
                           const float* __restrict__ scale, const float* __restrict__ shift,
                           bf16* __restrict__ hb) {
  long i = (long)blockIdx.x * 256 + threadIdx.x;
  if (i >= (long)cN * cF) return;
  int c = (int)(i % cF);
  float v = h[i] + scale[c] * agg[i] + shift[c];
  v = fmaxf(v, 0.f);
  h[i] = v; hb[i] = (bf16)v;
}

__global__ void k_pool(const float* __restrict__ h, const int* __restrict__ batch,
                       float* __restrict__ pooled, float* __restrict__ cnt) {
  long i = (long)blockIdx.x * 256 + threadIdx.x;
  if (i >= (long)cN * cF) return;
  int n = (int)(i / cF), f = (int)(i % cF);
  int g = batch[n];
  atomAddF(&pooled[(long)g * cF + f], h[i]);
  if (f == 0) atomAddF(&cnt[g], 1.f);
}

// per-graph head: shifted-softplus(fc) -> dot with out_w
__global__ __launch_bounds__(256)
void k_head(const float* __restrict__ pooled, const float* __restrict__ cnt,
            const float* __restrict__ fcw, const float* __restrict__ fcb,
            const float* __restrict__ ow, const float* __restrict__ ob,
            float* __restrict__ out) {
  __shared__ float red[256];
  int g = blockIdx.x, f = threadIdx.x;
  float c = fmaxf(cnt[g], 1.f);
  float acc = fcb[f];
  for (int k = 0; k < cF; ++k) acc += (pooled[(long)g * cF + k] / c) * fcw[(long)k * cF + f];
  float feat = dev_softplus(acc) - 0.6931471805599453f;
  red[f] = feat * ow[f];
  __syncthreads();
  for (int s = 128; s > 0; s >>= 1) { if (f < s) red[f] += red[f + s]; __syncthreads(); }
  if (f == 0) out[g] = red[0] + ob[0];
}

// ---------------------------------------------------------------------------
extern "C" void kernel_launch(void* const* d_in, const int* in_sizes, int n_in,
                              void* d_out, int out_size, void* d_ws, size_t ws_size,
                              hipStream_t stream) {
  const float* x      = (const float*)d_in[0];
  const int*   eix    = (const int*)  d_in[1];
  const float* ea     = (const float*)d_in[2];
  const int*   iix    = (const int*)  d_in[3];
  const float* iea    = (const float*)d_in[4];
  const int*   batch  = (const int*)  d_in[5];
  const float* atom_w = (const float*)d_in[6];
  const float* atom_b = (const float*)d_in[7];
  const float* edge_w = (const float*)d_in[8];
  const float* edge_b = (const float*)d_in[9];
  const float* inf_w  = (const float*)d_in[10];
  const float* inf_b  = (const float*)d_in[11];
  const float* ibn_g  = (const float*)d_in[12];
  const float* ibn_b  = (const float*)d_in[13];
  const float* w1f    = (const float*)d_in[14];
  const float* b1f    = (const float*)d_in[15];
  const float* w2f    = (const float*)d_in[16];
  const float* b2f    = (const float*)d_in[17];
  const float* w1     = (const float*)d_in[18];
  const float* b1     = (const float*)d_in[19];
  const float* w2     = (const float*)d_in[20];
  const float* b2     = (const float*)d_in[21];
  const float* bn_g   = (const float*)d_in[22];
  const float* bn_b   = (const float*)d_in[23];
  const float* bni_g  = (const float*)d_in[24];
  const float* bni_b  = (const float*)d_in[25];
  const float* fc_w   = (const float*)d_in[26];
  const float* fc_b   = (const float*)d_in[27];
  const float* out_w  = (const float*)d_in[28];
  const float* out_b  = (const float*)d_in[29];
  float* out = (float*)d_out;

  // ---- workspace carve-out ----
  char* ws = (char*)d_ws;
  size_t cur = 0;
  auto alloc = [&](size_t bytes) -> char* {
    cur = (cur + 255) & ~(size_t)255;
    char* p = ws + cur; cur += bytes; return p;
  };
  float* h     = (float*)alloc((size_t)cN * cF * 4);
  bf16*  hb    = (bf16*) alloc((size_t)cN * cF * 2);
  // efb: consumed as [cET x 256]; local-GEMM spill rows [cE, cEP) lie inside
  // this range and are overwritten by k_bn_apply_bf before any consumer runs.
  bf16*  efb   = (bf16*) alloc((size_t)cET * cF * 2);
  bf16*  z     = (bf16*) alloc((size_t)cETP * cK3 * 2);        // 460 MB
  bf16*  hid   = (bf16*) alloc((size_t)cETP * cF * 2);
  float* gateF = (float*)alloc((size_t)cETP * cF * 4);
  float* msgF  = (float*)alloc((size_t)cETP * cF * 4);
  float* agg   = (float*)alloc((size_t)cN * cF * 4);
  float* ssum  = (float*)alloc(cF * 4);
  float* ssq   = (float*)alloc(cF * 4);
  float* bsc   = (float*)alloc(cF * 4);
  float* bsh   = (float*)alloc(cF * 4);
  float* pooled= (float*)alloc((size_t)cG * cF * 4);
  float* cnt   = (float*)alloc((size_t)cG * 4);
  bf16*  ewp   = (bf16*) alloc((size_t)cF * cF * 2);
  bf16*  iwp   = (bf16*) alloc((size_t)cIB * cF * 2);
  bf16*  w1fp  = (bf16*) alloc((size_t)cL * cK3 * cF * 2);
  bf16*  w1p   = (bf16*) alloc((size_t)cL * cK3 * cF * 2);
  bf16*  w2fp  = (bf16*) alloc((size_t)cL * cF * cF * 2);
  bf16*  w2p   = (bf16*) alloc((size_t)cL * cF * cF * 2);
  // overlays inside the z region (consumed before the first z gather,
  // stream-ordered): RBF activation matrices for the two feature GEMMs.
  bf16*  rbfL  = z;                                             // [cEP  x 256] bf16 (102.5 MB)
  bf16*  rbfI  = (bf16*)((char*)z + 200000000);                 // [cEIP x  64] bf16 (12.8 MB)
  float* einfF = gateF;                                         // [cEIP x 256] f32 overlay (102.5 MB)

  auto cd = [](long a, long b) -> unsigned { return (unsigned)((a + b - 1) / b); };

  // ---- weight packing ----
  hipLaunchKernelGGL(k_pack_w, dim3(cd((long)cF  * cF, 256)), dim3(256), 0, stream, edge_w, ewp, cF);
  hipLaunchKernelGGL(k_pack_w, dim3(cd((long)cIB * cF, 256)), dim3(256), 0, stream, inf_w,  iwp, cIB);
  for (int l = 0; l < cL; ++l) {
    hipLaunchKernelGGL(k_pack_w, dim3(cd((long)cK3 * cF, 256)), dim3(256), 0, stream,
                       w1f + (size_t)l * cK3 * cF, w1fp + (size_t)l * cK3 * cF, cK3);
    hipLaunchKernelGGL(k_pack_w, dim3(cd((long)cK3 * cF, 256)), dim3(256), 0, stream,
                       w1  + (size_t)l * cK3 * cF, w1p  + (size_t)l * cK3 * cF, cK3);
    hipLaunchKernelGGL(k_pack_w, dim3(cd((long)cF * cF, 256)), dim3(256), 0, stream,
                       w2f + (size_t)l * cF * cF, w2fp + (size_t)l * cF * cF, cF);
    hipLaunchKernelGGL(k_pack_w, dim3(cd((long)cF * cF, 256)), dim3(256), 0, stream,
                       w2  + (size_t)l * cF * cF, w2p  + (size_t)l * cF * cF, cF);
  }

  // ---- atom embedding ----
  hipLaunchKernelGGL(k_embed, dim3(cd((long)cN * cF, 256)), dim3(256), 0, stream,
                     x, atom_w, atom_b, h, hb);

  // ---- local edge features: SiLU(RBF @ edge_w + b) -> ef_bf rows [0,E) ----
  hipLaunchKernelGGL(k_rbf_local, dim3(cd((long)cE * cF, 256)), dim3(256), 0, stream, ea, rbfL);
  hipLaunchKernelGGL(k_gemm_bf16, dim3(cEP / 128), dim3(256), 0, stream,
                     rbfL, ewp, edge_b, (float*)nullptr, efb, cF, /*silu*/1);

  // ---- infinite edge features: BN(softplus(RBF @ inf_w + b)) -> ef_bf rows [E,ET) ----
  hipLaunchKernelGGL(k_rbf_inf, dim3(cd((long)cEI * cIB, 256)), dim3(256), 0, stream, iea, rbfI);
  hipLaunchKernelGGL(k_gemm_bf16, dim3(cEIP / 128), dim3(256), 0, stream,
                     rbfI, iwp, inf_b, einfF, (bf16*)nullptr, cIB, /*softplus*/2);
  hipLaunchKernelGGL(k_zero, dim3(2), dim3(256), 0, stream, ssum, (long)cF);
  hipLaunchKernelGGL(k_zero, dim3(2), dim3(256), 0, stream, ssq,  (long)cF);
  hipLaunchKernelGGL(k_colstats, dim3(cd(cEI, 2048)), dim3(256), 0, stream, einfF, cEI, 2048, ssum, ssq);
  hipLaunchKernelGGL(k_bn_finalize, dim3(1), dim3(256), 0, stream, ssum, ssq, ibn_g, ibn_b, cEI, bsc, bsh);
  hipLaunchKernelGGL(k_bn_apply_bf, dim3(cd((long)cEI * cF, 256)), dim3(256), 0, stream,
                     einfF, bsc, bsh, efb + (size_t)cE * cF, (long)cEI);

  // ---- conv layers ----
  for (int l = 0; l < cL; ++l) {
    hipLaunchKernelGGL(k_gather_z, dim3(cd((long)cET * 96, 256)), dim3(256), 0, stream,
                       hb, efb, eix, iix, z);
    // gate MLP: hid = silu(z@w1f+b1f); gate = hid@w2f+b2f
    hipLaunchKernelGGL(k_gemm_bf16, dim3(cETP / 128), dim3(256), 0, stream,
                       z, w1fp + (size_t)l * cK3 * cF, b1f + (size_t)l * cF,
                       (float*)nullptr, hid, cK3, 1);
    hipLaunchKernelGGL(k_gemm_bf16, dim3(cETP / 128), dim3(256), 0, stream,
                       hid, w2fp + (size_t)l * cF * cF, b2f + (size_t)l * cF,
                       gateF, (bf16*)nullptr, cF, 0);
    // message MLP: hid = silu(z@w1+b1); msg = hid@w2+b2
    hipLaunchKernelGGL(k_gemm_bf16, dim3(cETP / 128), dim3(256), 0, stream,
                       z, w1p + (size_t)l * cK3 * cF, b1 + (size_t)l * cF,
                       (float*)nullptr, hid, cK3, 1);
    hipLaunchKernelGGL(k_gemm_bf16, dim3(cETP / 128), dim3(256), 0, stream,
                       hid, w2p + (size_t)l * cF * cF, b2 + (size_t)l * cF,
                       msgF, (bf16*)nullptr, cF, 0);
    // BN over edges of gate pre-activation
    hipLaunchKernelGGL(k_zero, dim3(2), dim3(256), 0, stream, ssum, (long)cF);
    hipLaunchKernelGGL(k_zero, dim3(2), dim3(256), 0, stream, ssq,  (long)cF);
    hipLaunchKernelGGL(k_colstats, dim3(cd(cET, 2048)), dim3(256), 0, stream, gateF, cET, 2048, ssum, ssq);
    hipLaunchKernelGGL(k_bn_finalize, dim3(1), dim3(256), 0, stream,
                       ssum, ssq, bni_g + (size_t)l * cF, bni_b + (size_t)l * cF, cET, bsc, bsh);
    // aggregate sigmoid(BN(gate))*msg into agg[tgt]
    hipLaunchKernelGGL(k_zero, dim3(cd((long)cN * cF, 256)), dim3(256), 0, stream, agg, (long)cN * cF);
    hipLaunchKernelGGL(k_gate_agg, dim3(cd((long)cET * cF, 256)), dim3(256), 0, stream,
                       gateF, msgF, bsc, bsh, eix, iix, agg);
    // node BN + relu residual
    hipLaunchKernelGGL(k_zero, dim3(2), dim3(256), 0, stream, ssum, (long)cF);
    hipLaunchKernelGGL(k_zero, dim3(2), dim3(256), 0, stream, ssq,  (long)cF);
    hipLaunchKernelGGL(k_colstats, dim3(cd(cN, 512)), dim3(256), 0, stream, agg, cN, 512, ssum, ssq);
    hipLaunchKernelGGL(k_bn_finalize, dim3(1), dim3(256), 0, stream,
                       ssum, ssq, bn_g + (size_t)l * cF, bn_b + (size_t)l * cF, cN, bsc, bsh);
    hipLaunchKernelGGL(k_update_h, dim3(cd((long)cN * cF, 256)), dim3(256), 0, stream,
                       h, agg, bsc, bsh, hb);
  }

  // ---- pool + head ----
  hipLaunchKernelGGL(k_zero, dim3(cd((long)cG * cF, 256)), dim3(256), 0, stream, pooled, (long)cG * cF);
  hipLaunchKernelGGL(k_zero, dim3(1), dim3(256), 0, stream, cnt, (long)cG);
  hipLaunchKernelGGL(k_pool, dim3(cd((long)cN * cF, 256)), dim3(256), 0, stream, h, batch, pooled, cnt);
  hipLaunchKernelGGL(k_head, dim3(cG), dim3(256), 0, stream,
                     pooled, cnt, fc_w, fc_b, out_w, out_b, out);
}
